// StandardMoELayer_9405978378748
// MI455X (gfx1250) — matrix operand order repaired and checked by reading, not compiled, not run on hardware
//
#include <hip/hip_runtime.h>
#include <hip/hip_bf16.h>
#include <cmath>

// ---------------- problem constants (from reference) ----------------
constexpr int kE = 8;     // experts
constexpr int kC = 1024;  // d_model
constexpr int kF = 2048;  // d_ffn
constexpr int kN = 2048;  // tokens (2*1024)
constexpr float kEPS = 1e-5f;

// ---------------- vector types for WMMA ----------------
typedef __attribute__((ext_vector_type(16))) __bf16 v16bf;
typedef __attribute__((ext_vector_type(8)))  __bf16 v8bf;
typedef __attribute__((ext_vector_type(4)))  __bf16 v4bf;
typedef __attribute__((ext_vector_type(8)))  float  v8f;

// ---------------- GEMM tiling ----------------
constexpr int TILE_M  = 128;   // 8 waves x 16 rows
constexpr int TILE_N  = 128;   // 8 WMMA col-tiles per wave
constexpr int KCHUNK  = 64;    // K depth staged per LDS buffer (2 wmma k-sub-steps)
constexpr int BSTRIDE = 144;   // padded bytes per B column (128 data + 16 pad; 16B aligned, conflict-free)
constexpr int LDSB    = TILE_N * BSTRIDE;  // 18432 bytes per buffer

// ---------------- workspace layout (bytes) ----------------
constexpr size_t OFF_XB   = 0;                                   // [N][C] bf16 : 4 MiB
constexpr size_t OFF_COMB = OFF_XB   + (size_t)kN * kC * 2;      // [N][E] f32  : 64 KiB
constexpr size_t OFF_W1B  = OFF_COMB + (size_t)kN * kE * 4;      // [E][F][C] bf16 : 32 MiB
constexpr size_t OFF_W2B  = OFF_W1B  + (size_t)kE * kF * kC * 2; // [E][C][F] bf16 : 32 MiB
constexpr size_t OFF_H    = OFF_W2B  + (size_t)kE * kC * kF * 2; // [N][F] bf16 : 8 MiB (one expert at a time)

// =====================================================================
// Kernel 1: router — LayerNorm + gate + softmax + top-2 combine weights.
// One wave32 per token. Also emits xb = bf16(x) (pre-LN x feeds the MLP).
// =====================================================================
__global__ __launch_bounds__(256) void router_kernel(
    const float* __restrict__ x, const float* __restrict__ gamma,
    const float* __restrict__ beta, const float* __restrict__ gw,
    __bf16* __restrict__ xb, float* __restrict__ comb)
{
  const int lane  = threadIdx.x & 31;
  const int token = blockIdx.x * 8 + (threadIdx.x >> 5);
  const float* row = x + (size_t)token * kC;

  float xv[32];
  float sum = 0.f, ssq = 0.f;
#pragma unroll
  for (int i = 0; i < 32; ++i) {
    const int c = lane + 32 * i;
    const float v = row[c];
    xv[i] = v; sum += v; ssq += v * v;
    xb[(size_t)token * kC + c] = (__bf16)v;   // GEMMs consume raw x
  }
#pragma unroll
  for (int m = 16; m >= 1; m >>= 1) {
    sum += __shfl_xor(sum, m, 32);
    ssq += __shfl_xor(ssq, m, 32);
  }
  const float mu   = sum * (1.0f / kC);
  const float var  = ssq * (1.0f / kC) - mu * mu;   // biased, torch-style
  const float rstd = rsqrtf(var + kEPS);

  float lg[kE] = {};
#pragma unroll
  for (int i = 0; i < 32; ++i) {
    const int c = lane + 32 * i;
    const float rn = (xv[i] - mu) * rstd * gamma[c] + beta[c];
#pragma unroll
    for (int e = 0; e < kE; ++e) lg[e] += rn * gw[e * kC + c];
  }
#pragma unroll
  for (int e = 0; e < kE; ++e)
#pragma unroll
    for (int m = 16; m >= 1; m >>= 1) lg[e] += __shfl_xor(lg[e], m, 32);

  if (lane == 0) {
    float mx = lg[0];
#pragma unroll
    for (int e = 1; e < kE; ++e) mx = fmaxf(mx, lg[e]);
    float p[kE], sm = 0.f;
#pragma unroll
    for (int e = 0; e < kE; ++e) { p[e] = expf(lg[e] - mx); sm += p[e]; }
    const float inv = 1.0f / sm;
#pragma unroll
    for (int e = 0; e < kE; ++e) p[e] *= inv;
    int i1 = 0;
#pragma unroll
    for (int e = 1; e < kE; ++e) if (p[e] > p[i1]) i1 = e;   // first max on tie (jax top_k)
    int i2 = (i1 == 0) ? 1 : 0;
#pragma unroll
    for (int e = 0; e < kE; ++e) if (e != i1 && p[e] > p[i2]) i2 = e;
#pragma unroll
    for (int e = 0; e < kE; ++e)
      comb[(size_t)token * kE + e] = (e == i1) ? p[i1] : ((e == i2) ? p[i2] : 0.0f);
  }
}

// =====================================================================
// Kernel 2: f32 -> bf16 streaming convert (weights), float4 granularity.
// =====================================================================
__global__ __launch_bounds__(256) void cvt_bf16_kernel(
    const float* __restrict__ src, __bf16* __restrict__ dst, int n4)
{
  int i = blockIdx.x * blockDim.x + threadIdx.x;
  const int stride = gridDim.x * blockDim.x;
  for (; i < n4; i += stride) {
    const float4 v = ((const float4*)src)[i];
    v4bf o;
    o[0] = (__bf16)v.x; o[1] = (__bf16)v.y; o[2] = (__bf16)v.z; o[3] = (__bf16)v.w;
    ((v4bf*)dst)[i] = o;
  }
}

// ---------------- fragment helpers (ISA 7.12.2 16-bit layouts) ----------------
// A (16x32): lane = 16*hi + m ; halves [0..7] = K[hi*8 .. +7], [8..15] = K[16+hi*8 .. +7]
__device__ __forceinline__ v16bf load_a_frag(const __bf16* __restrict__ rowp, int k0)
{
  v16bf a;
  const v8bf a0 = *(const v8bf*)(rowp + k0);
  const v8bf a1 = *(const v8bf*)(rowp + k0 + 16);
#pragma unroll
  for (int i = 0; i < 8; ++i) { a[i] = a0[i]; a[8 + i] = a1[i]; }
  return a;
}

// B fragment from padded LDS tile: idx = u*8 + t (u = k-sub-step, t = col-tile)
__device__ __forceinline__ v16bf load_b_lds(const char* bb, int idx, int ln, int hi)
{
  const int u = idx >> 3, t = idx & 7;
  const __bf16* p = (const __bf16*)(bb + (t * 16 + ln) * BSTRIDE + u * 64 + hi * 16);
  v16bf b;
  const v8bf b0 = *(const v8bf*)p;
  const v8bf b1 = *(const v8bf*)(p + 16);
#pragma unroll
  for (int i = 0; i < 8; ++i) { b[i] = b0[i]; b[8 + i] = b1[i]; }
  return b;
}

// async stage of one 64-byte slice of the B tile: LDS[l..l+63] = G[g..g+63]
// (IOFFSET applies to both the LDS and global address per ISA 15.18.3)
__device__ __forceinline__ void stage64_async(unsigned l, const __bf16* g)
{
  asm volatile("global_load_async_to_lds_b128 %0, %1, off"           :: "v"(l), "v"(g) : "memory");
  asm volatile("global_load_async_to_lds_b128 %0, %1, off offset:16" :: "v"(l), "v"(g) : "memory");
  asm volatile("global_load_async_to_lds_b128 %0, %1, off offset:32" :: "v"(l), "v"(g) : "memory");
  asm volatile("global_load_async_to_lds_b128 %0, %1, off offset:48" :: "v"(l), "v"(g) : "memory");
}
__device__ __forceinline__ void wait_async0()
{
  asm volatile("s_wait_asynccnt 0x0" ::: "memory");
}

// =====================================================================
// Shared GEMM mainloop: C[128x128] tile, B staged to LDS by async DMA
// (double-buffered, 64-deep K chunks). Two live B fragments (bA even tiles,
// bB odd tiles), each reloaded two tiles ahead so every ds pair has ~2 WMMAs
// of latency cover and waits can relax to dscnt<=1.
// =====================================================================
template <int KDIM>
__device__ __forceinline__ void gemm_mainloop(
    const __bf16* __restrict__ arow,   // A row base (+hi*8), stride KDIM
    const __bf16* __restrict__ sg,     // this thread's staging source (col/half resolved)
    unsigned soff,                     // this thread's LDS staging offset (bytes)
    char (&ldsb)[2][LDSB], int ln, int hi, v8f acc[8])
{
  const unsigned lbase0 = (unsigned)(size_t)&ldsb[0][0];
  const unsigned lbase1 = (unsigned)(size_t)&ldsb[1][0];

  stage64_async(lbase0 + soff, sg);                 // prologue: chunk 0 -> buf 0

  constexpr int steps = KDIM / KCHUNK;
  for (int s = 0; s < steps; ++s) {
    wait_async0();
    __syncthreads();                                // buf[s&1] ready for all waves
    if (s + 1 < steps)
      stage64_async(((s + 1) & 1 ? lbase1 : lbase0) + soff, sg + (s + 1) * KCHUNK);

    const v16bf a0 = load_a_frag(arow, s * KCHUNK);
    const v16bf a1 = load_a_frag(arow, s * KCHUNK + 32);
    const char* bb = ldsb[s & 1];

    v16bf bA = load_b_lds(bb, 0, ln, hi);
    v16bf bB = load_b_lds(bb, 1, ln, hi);
#pragma unroll
    for (int i = 0; i < 16; i += 2) {               // u = i>>3, t = i&7
      acc[i & 7] = __builtin_amdgcn_wmma_f32_16x16x32_bf16(
          false, (i < 8) ? a0 : a1, false, bA, (short)0, acc[i & 7], false, false);
      if (i + 2 < 16) bA = load_b_lds(bb, i + 2, ln, hi);

      const int j = i + 1;
      acc[j & 7] = __builtin_amdgcn_wmma_f32_16x16x32_bf16(
          false, (j < 8) ? a0 : a1, false, bB, (short)0, acc[j & 7], false, false);
      if (j + 2 < 16) bB = load_b_lds(bb, j + 2, ln, hi);
    }
    __syncthreads();                                // all reads of buf[s&1] done before overwrite
  }
}

// =====================================================================
// Kernel 3: GEMM1 + GELU + combine-fold, one expert.
//   h'[n,f] = comb[n,e] * gelu( sum_c xb[n,c] * w1[e,f,c] )
// =====================================================================
__global__ __launch_bounds__(256) void gemm1_kernel(
    const __bf16* __restrict__ xb, const __bf16* __restrict__ w1b /* expert base, [F][C] */,
    const float* __restrict__ comb, __bf16* __restrict__ h, int expert)
{
  __shared__ __align__(16) char ldsb[2][LDSB];

  const int tid  = threadIdx.x;
  const int wave = tid >> 5, lane = tid & 31;
  const int ln = lane & 15, hi = lane >> 4;
  const int m0 = blockIdx.x * TILE_M + wave * 16;
  const int n0 = blockIdx.y * TILE_N;

  // staging role: 2 threads per B column, 64 bytes (32 K elems) each
  const int scol = tid >> 1, shalf = tid & 1;
  const __bf16* sg = w1b + (size_t)(n0 + scol) * kC + shalf * 32;
  const unsigned soff = (unsigned)(scol * BSTRIDE + shalf * 64);

  v8f acc[8] = {};
  const __bf16* arow = xb + (size_t)(m0 + ln) * kC + hi * 8;

  gemm_mainloop<kC>(arow, sg, soff, ldsb, ln, hi, acc);

  // per-row combine weight (C/D row = j + 8*hi)
  float scl[8];
#pragma unroll
  for (int j = 0; j < 8; ++j)
    scl[j] = comb[(size_t)(m0 + j + 8 * hi) * kE + expert];

#pragma unroll
  for (int t = 0; t < 8; ++t) {
    const int col = n0 + t * 16 + ln;
#pragma unroll
    for (int j = 0; j < 8; ++j) {
      float v = acc[t][j];
      v = 0.5f * v * (1.0f + erff(v * 0.70710678118654752f));  // exact-erf GELU
      v *= scl[j];
      h[(size_t)(m0 + j + 8 * hi) * kF + col] = (__bf16)v;
    }
  }
}

// =====================================================================
// Kernel 4: GEMM2 accumulate into out, one expert.
//   out[n,c] (+)= sum_f h'[n,f] * w2[e,c,f]     K = F = 2048
// =====================================================================
__global__ __launch_bounds__(256) void gemm2_kernel(
    const __bf16* __restrict__ h, const __bf16* __restrict__ w2b /* expert base, [C][F] */,
    float* __restrict__ out, int add_flag)
{
  __shared__ __align__(16) char ldsb[2][LDSB];

  const int tid  = threadIdx.x;
  const int wave = tid >> 5, lane = tid & 31;
  const int ln = lane & 15, hi = lane >> 4;
  const int m0 = blockIdx.x * TILE_M + wave * 16;
  const int n0 = blockIdx.y * TILE_N;

  const int scol = tid >> 1, shalf = tid & 1;
  const __bf16* sg = w2b + (size_t)(n0 + scol) * kF + shalf * 32;
  const unsigned soff = (unsigned)(scol * BSTRIDE + shalf * 64);

  v8f acc[8] = {};
  const __bf16* arow = h + (size_t)(m0 + ln) * kF + hi * 8;

  gemm_mainloop<kF>(arow, sg, soff, ldsb, ln, hi, acc);

#pragma unroll
  for (int t = 0; t < 8; ++t) {
    const int col = n0 + t * 16 + ln;
#pragma unroll
    for (int j = 0; j < 8; ++j) {
      const size_t idx = (size_t)(m0 + j + 8 * hi) * kC + col;
      float v = acc[t][j];
      if (add_flag) v += out[idx];
      out[idx] = v;
    }
  }
}

// =====================================================================
extern "C" void kernel_launch(void* const* d_in, const int* in_sizes, int n_in,
                              void* d_out, int out_size, void* d_ws, size_t ws_size,
                              hipStream_t stream) {
  (void)in_sizes; (void)n_in; (void)out_size; (void)ws_size;
  const float* x     = (const float*)d_in[0];
  const float* gamma = (const float*)d_in[1];
  const float* beta  = (const float*)d_in[2];
  const float* gw    = (const float*)d_in[3];
  const float* w1    = (const float*)d_in[4];
  const float* w2    = (const float*)d_in[5];
  float* out = (float*)d_out;

  char* ws = (char*)d_ws;
  __bf16* xb   = (__bf16*)(ws + OFF_XB);
  float*  comb = (float*)(ws + OFF_COMB);
  __bf16* w1b  = (__bf16*)(ws + OFF_W1B);
  __bf16* w2b  = (__bf16*)(ws + OFF_W2B);
  __bf16* h    = (__bf16*)(ws + OFF_H);

  // 1) router + x->bf16
  router_kernel<<<kN / 8, 256, 0, stream>>>(x, gamma, beta, gw, xb, comb);

  // 2) weight conversion (16.78M elems each -> 4.19M float4)
  const int n4 = (kE * kF * kC) / 4;
  cvt_bf16_kernel<<<4096, 256, 0, stream>>>(w1, w1b, n4);
  cvt_bf16_kernel<<<4096, 256, 0, stream>>>(w2, w2b, n4);

  // 3) per-expert fused MLP: h' = comb .* gelu(x @ w1^T); out += h' @ w2^T
  for (int e = 0; e < kE; ++e) {
    const __bf16* w1e = w1b + (size_t)e * kF * kC;
    const __bf16* w2e = w2b + (size_t)e * kC * kF;
    gemm1_kernel<<<dim3(kN / TILE_M, kF / TILE_N), 256, 0, stream>>>(xb, w1e, comb, h, e);
    gemm2_kernel<<<dim3(kN / TILE_M, kC / TILE_N), 256, 0, stream>>>(h, w2e, out, e == 0 ? 0 : 1);
  }
}
